// Attention_6622839571380
// MI455X (gfx1250) — matrix operand order repaired and checked by reading, not compiled, not run on hardware
//
#include <hip/hip_runtime.h>
#include <hip/hip_bf16.h>

// ---------------------------------------------------------------------------
// MI455X (gfx1250, wave32) multi-head attention:
//   q/k/v = x @ W{q,k,v}^T + b ; softmax(q k^T / 8) v ; @ Wo^T + bo
// bf16 WMMA (v_wmma_f32_16x16x32_bf16) with f32 accumulation everywhere.
// Attention K/V tiles staged to LDS with gfx1250 async copies
// (global_load_async_to_lds_b128 + s_wait_asynccnt), double-buffered.
// ---------------------------------------------------------------------------

typedef __attribute__((ext_vector_type(16))) __bf16 v16bf;
typedef __attribute__((ext_vector_type(8)))  float  v8f;

struct __align__(16) U4 { unsigned int a, b, c, d; };
struct __align__(16) F4 { float x, y, z, w; };

union AFrag { v16bf v; U4 q[2]; };          // 32 bytes: one 16x32 bf16 operand slice per lane
union Pack8 { U4 q; __bf16 h[8]; };         // 8 bf16 packed for b128 stores

constexpr int Bc  = 16;     // batch
constexpr int Sc_ = 1024;   // sequence
constexpr int Dc  = 768;    // model dim
constexpr int Hc  = 12;     // heads
constexpr int HDc = 64;     // head dim

__device__ __forceinline__ v8f zero8() {
  v8f z = {0.f, 0.f, 0.f, 0.f, 0.f, 0.f, 0.f, 0.f};
  return z;
}

__device__ __forceinline__ v8f wmma_bf16(v16bf a, v16bf b, v8f c) {
  // (neg_a, A, neg_b, B, c_mod, C, reuse_a, reuse_b)
  return __builtin_amdgcn_wmma_f32_16x16x32_bf16(false, a, false, b, (short)0, c,
                                                 false, false);
}

// Async copy 16 bytes/lane from global to LDS (ASYNCcnt-tracked, ISA §15.18.3).
__device__ __forceinline__ void async_copy_b128(unsigned lds_off, const __bf16* g) {
  asm volatile("global_load_async_to_lds_b128 %0, %1, off"
               :: "v"(lds_off), "v"((unsigned long long)(uintptr_t)g)
               : "memory");
}
__device__ __forceinline__ void wait_async0() {
  asm volatile("s_wait_asynccnt 0x0" ::: "memory");
}

// ---------------------------------------------------------------------------
// Kernel 1: convert the four 768x768 f32 weight matrices to bf16 in ws.
// ---------------------------------------------------------------------------
__global__ void cvt_w_kernel(const float* __restrict__ wq, const float* __restrict__ wk,
                             const float* __restrict__ wv, const float* __restrict__ wo,
                             __bf16* __restrict__ dst) {
  const int n = Dc * Dc;
  const int w = blockIdx.y;
  const float* src = (w == 0) ? wq : (w == 1) ? wk : (w == 2) ? wv : wo;
  __bf16* d = dst + (size_t)w * n;
  for (int e = blockIdx.x * blockDim.x + threadIdx.x; e < n;
       e += gridDim.x * blockDim.x)
    d[e] = (__bf16)src[e];
}

// ---------------------------------------------------------------------------
// Kernel 2: fused QKV projection. One wave -> 64x64 output tile.
//   grid = (M/64 = 256, heads = 12, proj = 3), block = 32 (one wave).
// A = x (f32 -> bf16 in-register), B = W rows (K-contiguous == WMMA B layout).
// Q/K stored [b,h,s,d] bf16; V stored transposed [b,h,d,s] bf16 with packed
// b128 stores (C-layout gives each lane 8 consecutive-M values at fixed N).
// ---------------------------------------------------------------------------
__global__ void __launch_bounds__(32)
qkv_kernel(const float* __restrict__ x,
           const __bf16* __restrict__ wqb, const __bf16* __restrict__ wkb,
           const __bf16* __restrict__ wvb,
           const float* __restrict__ bq, const float* __restrict__ bk,
           const float* __restrict__ bv,
           __bf16* __restrict__ qo, __bf16* __restrict__ ko,
           __bf16* __restrict__ vto) {
  const int lane = threadIdx.x;
  const int half = lane >> 4, l16 = lane & 15;
  const int m0 = blockIdx.x * 64;           // row in [B*S]
  const int h  = blockIdx.y;                // head -> n0 = h*64
  const int z  = blockIdx.z;                // 0=q 1=k 2=v
  const __bf16* wb   = (z == 0) ? wqb : (z == 1) ? wkb : wvb;
  const float*  bias = (z == 0) ? bq  : (z == 1) ? bk  : bv;

  v8f C[4][4];
  #pragma unroll
  for (int i = 0; i < 4; ++i)
    #pragma unroll
    for (int j = 0; j < 4; ++j) C[i][j] = zero8();

  for (int k0 = 0; k0 < Dc; k0 += 32) {
    AFrag A[4];
    #pragma unroll
    for (int i = 0; i < 4; ++i) {
      // A layout: lane row = m, K chunks {half*8, 16+half*8}, 8 elems each
      const float* xr = x + (size_t)(m0 + i * 16 + l16) * Dc + k0 + half * 8;
      F4 f0 = *(const F4*)(xr);
      F4 f1 = *(const F4*)(xr + 4);
      F4 f2 = *(const F4*)(xr + 16);
      F4 f3 = *(const F4*)(xr + 20);
      v16bf av;
      av[0]  = (__bf16)f0.x; av[1]  = (__bf16)f0.y; av[2]  = (__bf16)f0.z; av[3]  = (__bf16)f0.w;
      av[4]  = (__bf16)f1.x; av[5]  = (__bf16)f1.y; av[6]  = (__bf16)f1.z; av[7]  = (__bf16)f1.w;
      av[8]  = (__bf16)f2.x; av[9]  = (__bf16)f2.y; av[10] = (__bf16)f2.z; av[11] = (__bf16)f2.w;
      av[12] = (__bf16)f3.x; av[13] = (__bf16)f3.y; av[14] = (__bf16)f3.z; av[15] = (__bf16)f3.w;
      A[i].v = av;
    }
    #pragma unroll
    for (int j = 0; j < 4; ++j) {
      // B layout: lane = column n = row n of W, K chunks {half*8, 16+half*8}, 8 elems each
      AFrag Bf;
      const __bf16* wr = wb + (size_t)(h * 64 + j * 16 + l16) * Dc + k0 + half * 8;
      Bf.q[0] = *(const U4*)(wr);
      Bf.q[1] = *(const U4*)(wr + 16);
      #pragma unroll
      for (int i = 0; i < 4; ++i) C[i][j] = wmma_bf16(A[i].v, Bf.v, C[i][j]);
    }
  }

  const int b  = m0 / Sc_;
  const int s0 = m0 % Sc_;
  #pragma unroll
  for (int j = 0; j < 4; ++j) {
    const int d = j * 16 + l16;
    const float bv_ = bias[h * 64 + d];
    if (z < 2) {
      __bf16* dst = (z == 0 ? qo : ko) + (size_t)(b * Hc + h) * Sc_ * HDc;
      #pragma unroll
      for (int i = 0; i < 4; ++i)
        #pragma unroll
        for (int p = 0; p < 8; ++p) {
          const int s = s0 + i * 16 + half * 8 + p;
          dst[(size_t)s * HDc + d] = (__bf16)(C[i][j][p] + bv_);
        }
    } else {
      // V transposed: [b,h,d,s]; 8 consecutive s per lane -> one b128 store
      __bf16* dst = vto + ((size_t)(b * Hc + h) * HDc + d) * Sc_;
      #pragma unroll
      for (int i = 0; i < 4; ++i) {
        Pack8 pk;
        #pragma unroll
        for (int p = 0; p < 8; ++p) pk.h[p] = (__bf16)(C[i][j][p] + bv_);
        const int s = s0 + i * 16 + half * 8;
        *(U4*)(dst + s) = pk.q;
      }
    }
  }
}

// ---------------------------------------------------------------------------
// Kernel 3: flash attention. grid = (B*H = 192, S/128 = 8), block = 128
// (4 waves). Each wave: 32 query rows x full head, streaming 64 kv rows/iter.
// K / V^T tiles are async-copied to LDS ONCE per workgroup (double-buffered,
// overlapping the DMA for tile t+1 with WMMA on tile t), removing the 4x
// redundant global reads. Scores in C-layout -> online softmax (shfl_xor
// inside 16-lane halves) -> P staged to LDS row-major -> PV WMMA.
// ---------------------------------------------------------------------------
__global__ void __launch_bounds__(128)
attn_kernel(const __bf16* __restrict__ qg, const __bf16* __restrict__ kg,
            const __bf16* __restrict__ vtg, __bf16* __restrict__ attn) {
  // 72 = 64 + pad: keeps every 16-byte fragment chunk aligned, breaks bank conflicts
  __shared__ __align__(16) __bf16 Kl[2][64][72];   // [buf][kv][d]
  __shared__ __align__(16) __bf16 Vl[2][64][72];   // [buf][d][kv]
  __shared__ __align__(16) __bf16 Pl[4][32][72];   // per-wave probs

  const int tid  = threadIdx.x;
  const int lane = tid & 31;
  const int wave = tid >> 5;
  const int half = lane >> 4, l16 = lane & 15;
  const int bh = blockIdx.x;
  const int b = bh / Hc, h = bh % Hc;
  const int q0 = blockIdx.y * 128 + wave * 32;     // sequence offset

  const __bf16* qp  = qg  + (size_t)bh * Sc_ * HDc;
  const __bf16* kp  = kg  + (size_t)bh * Sc_ * HDc;
  const __bf16* vtp = vtg + (size_t)bh * HDc * Sc_;

  // cooperative async stage of one 64x64 K tile + one 64x64 V^T tile:
  // 512 chunks of 16B per tile, 4 per thread.
  auto issue_tile = [&](int t, int buf) {
    #pragma unroll
    for (int c = 0; c < 4; ++c) {
      const int chunk = tid + c * 128;       // 0..511
      const int row   = chunk >> 3;          // 0..63
      const int col   = (chunk & 7) * 8;     // element offset, 16B granules
      async_copy_b128((unsigned)(uintptr_t)&Kl[buf][row][col],
                      kp + (size_t)(t * 64 + row) * HDc + col);
      async_copy_b128((unsigned)(uintptr_t)&Vl[buf][row][col],
                      vtp + (size_t)row * Sc_ + t * 64 + col);
    }
  };

  // persistent Q fragments: 2 m-tiles x 2 K-chunks of 32 over head dim 64
  AFrag Qf[2][2];
  #pragma unroll
  for (int i = 0; i < 2; ++i)
    #pragma unroll
    for (int kk = 0; kk < 2; ++kk) {
      const __bf16* r = qp + (size_t)(q0 + i * 16 + l16) * HDc + kk * 32 + half * 8;
      Qf[i][kk].q[0] = *(const U4*)(r);
      Qf[i][kk].q[1] = *(const U4*)(r + 16);
    }

  float mst[2][8], lst[2][8];
  v8f O[2][4];
  #pragma unroll
  for (int i = 0; i < 2; ++i) {
    #pragma unroll
    for (int p = 0; p < 8; ++p) { mst[i][p] = -1e30f; lst[i][p] = 0.f; }
    #pragma unroll
    for (int a = 0; a < 4; ++a) O[i][a] = zero8();
  }

  issue_tile(0, 0);

  for (int t = 0; t < Sc_ / 64; ++t) {
    const int buf = t & 1;
    wait_async0();        // this wave's pending tile (t) has landed in LDS
    __syncthreads();      // every wave's chunks have landed
    if (t + 1 < Sc_ / 64) issue_tile(t + 1, buf ^ 1);  // overlap DMA with math

    // ---- scores S = Q K^T (C-layout: VGPR p = row, lane = kv column) ----
    v8f Sc[2][4];
    #pragma unroll
    for (int j = 0; j < 4; ++j) {
      AFrag Kf[2];
      #pragma unroll
      for (int kk = 0; kk < 2; ++kk) {
        const __bf16* r = &Kl[buf][j * 16 + l16][kk * 32 + half * 8];
        Kf[kk].q[0] = *(const U4*)(r);
        Kf[kk].q[1] = *(const U4*)(r + 16);
      }
      #pragma unroll
      for (int i = 0; i < 2; ++i) {
        v8f c = zero8();
        c = wmma_bf16(Qf[i][0].v, Kf[0].v, c);
        c = wmma_bf16(Qf[i][1].v, Kf[1].v, c);
        Sc[i][j] = c;
      }
    }
    #pragma unroll
    for (int i = 0; i < 2; ++i)
      #pragma unroll
      for (int j = 0; j < 4; ++j) Sc[i][j] *= 0.125f;   // 1/sqrt(64)

    // ---- online softmax; masks < 16 keep shuffles inside each half ----
    #pragma unroll
    for (int i = 0; i < 2; ++i)
      #pragma unroll
      for (int p = 0; p < 8; ++p) {
        float rm = fmaxf(fmaxf(Sc[i][0][p], Sc[i][1][p]),
                         fmaxf(Sc[i][2][p], Sc[i][3][p]));
        rm = fmaxf(rm, __shfl_xor(rm, 1));
        rm = fmaxf(rm, __shfl_xor(rm, 2));
        rm = fmaxf(rm, __shfl_xor(rm, 4));
        rm = fmaxf(rm, __shfl_xor(rm, 8));
        const float mnew  = fmaxf(mst[i][p], rm);
        const float alpha = __expf(mst[i][p] - mnew);
        mst[i][p] = mnew;
        float ls = 0.f;
        #pragma unroll
        for (int j = 0; j < 4; ++j) {
          const float e = __expf(Sc[i][j][p] - mnew);
          Sc[i][j][p] = e;
          ls += e;
        }
        ls += __shfl_xor(ls, 1);
        ls += __shfl_xor(ls, 2);
        ls += __shfl_xor(ls, 4);
        ls += __shfl_xor(ls, 8);
        lst[i][p] = lst[i][p] * alpha + ls;
        #pragma unroll
        for (int a = 0; a < 4; ++a) O[i][a][p] *= alpha;  // rescale accumulator
      }

    // ---- stage P (bf16) row-major in this wave's LDS slice ----
    #pragma unroll
    for (int i = 0; i < 2; ++i)
      #pragma unroll
      for (int j = 0; j < 4; ++j)
        #pragma unroll
        for (int p = 0; p < 8; ++p)
          Pl[wave][i * 16 + half * 8 + p][j * 16 + l16] = (__bf16)Sc[i][j][p];

    // ---- O += P V : A = P (ds_load_b128), B = V^T rows from LDS ----
    #pragma unroll
    for (int kk = 0; kk < 2; ++kk) {
      AFrag Pf[2];
      #pragma unroll
      for (int i = 0; i < 2; ++i) {
        const __bf16* r = &Pl[wave][i * 16 + l16][kk * 32 + half * 8];
        Pf[i].q[0] = *(const U4*)(r);
        Pf[i].q[1] = *(const U4*)(r + 16);
      }
      #pragma unroll
      for (int a = 0; a < 4; ++a) {
        AFrag Vf;
        const __bf16* r = &Vl[buf][a * 16 + l16][kk * 32 + half * 8];
        Vf.q[0] = *(const U4*)(r);
        Vf.q[1] = *(const U4*)(r + 16);
        O[0][a] = wmma_bf16(Pf[0].v, Vf.v, O[0][a]);
        O[1][a] = wmma_bf16(Pf[1].v, Vf.v, O[1][a]);
      }
    }

    __syncthreads();      // all waves done reading buf before it is refilled
  }

  // ---- normalize and emit context as [b, s, h*64 + d] bf16 ----
  #pragma unroll
  for (int i = 0; i < 2; ++i)
    #pragma unroll
    for (int p = 0; p < 8; ++p) {
      const float inv = 1.f / lst[i][p];
      const int s = q0 + i * 16 + half * 8 + p;
      #pragma unroll
      for (int a = 0; a < 4; ++a)
        attn[(size_t)(b * Sc_ + s) * Dc + h * HDc + a * 16 + l16] =
            (__bf16)(O[i][a][p] * inv);
    }
}

// ---------------------------------------------------------------------------
// Kernel 4: output projection, out = attn @ Wo^T + bo (f32 out).
//   grid = (256, 12), block = 32 (one wave -> 64x64 tile).
// ---------------------------------------------------------------------------
__global__ void __launch_bounds__(32)
oproj_kernel(const __bf16* __restrict__ a, const __bf16* __restrict__ wob,
             const float* __restrict__ bo, float* __restrict__ out) {
  const int lane = threadIdx.x;
  const int half = lane >> 4, l16 = lane & 15;
  const int m0 = blockIdx.x * 64;
  const int n0 = blockIdx.y * 64;

  v8f C[4][4];
  #pragma unroll
  for (int i = 0; i < 4; ++i)
    #pragma unroll
    for (int j = 0; j < 4; ++j) C[i][j] = zero8();

  for (int k0 = 0; k0 < Dc; k0 += 32) {
    AFrag A[4];
    #pragma unroll
    for (int i = 0; i < 4; ++i) {
      const __bf16* r = a + (size_t)(m0 + i * 16 + l16) * Dc + k0 + half * 8;
      A[i].q[0] = *(const U4*)(r);
      A[i].q[1] = *(const U4*)(r + 16);
    }
    #pragma unroll
    for (int j = 0; j < 4; ++j) {
      AFrag Bf;
      const __bf16* wr = wob + (size_t)(n0 + j * 16 + l16) * Dc + k0 + half * 8;
      Bf.q[0] = *(const U4*)(wr);
      Bf.q[1] = *(const U4*)(wr + 16);
      #pragma unroll
      for (int i = 0; i < 4; ++i) C[i][j] = wmma_bf16(A[i].v, Bf.v, C[i][j]);
    }
  }

  #pragma unroll
  for (int j = 0; j < 4; ++j) {
    const float bv_ = bo[n0 + j * 16 + l16];
    #pragma unroll
    for (int i = 0; i < 4; ++i)
      #pragma unroll
      for (int p = 0; p < 8; ++p)
        out[(size_t)(m0 + i * 16 + half * 8 + p) * Dc + n0 + j * 16 + l16] =
            C[i][j][p] + bv_;
  }
}

// ---------------------------------------------------------------------------
extern "C" void kernel_launch(void* const* d_in, const int* in_sizes, int n_in,
                              void* d_out, int out_size, void* d_ws, size_t ws_size,
                              hipStream_t stream) {
  (void)in_sizes; (void)n_in; (void)out_size; (void)ws_size;

  const float* x    = (const float*)d_in[0];
  const float* wq_w = (const float*)d_in[1];
  const float* wq_b = (const float*)d_in[2];
  const float* wk_w = (const float*)d_in[3];
  const float* wk_b = (const float*)d_in[4];
  const float* wv_w = (const float*)d_in[5];
  const float* wv_b = (const float*)d_in[6];
  const float* wo_w = (const float*)d_in[7];
  const float* wo_b = (const float*)d_in[8];

  // workspace layout (bf16): 4 weights | Q | K | V^T | attn  (~100.5 MB)
  __bf16* ws = (__bf16*)d_ws;
  const size_t WN = (size_t)Dc * Dc;                 // 589824
  const size_t QN = (size_t)Bc * Sc_ * Dc;           // 12582912
  __bf16* wqb = ws;
  __bf16* wkb = wqb + WN;
  __bf16* wvb = wkb + WN;
  __bf16* wob = wvb + WN;
  __bf16* qws = wob + WN;
  __bf16* kws = qws + QN;
  __bf16* vtw = kws + QN;
  __bf16* atw = vtw + QN;

  cvt_w_kernel<<<dim3(576, 4), 256, 0, stream>>>(wq_w, wk_w, wv_w, wo_w, wqb);
  qkv_kernel<<<dim3(256, 12, 3), 32, 0, stream>>>(x, wqb, wkb, wvb,
                                                  wq_b, wk_b, wv_b,
                                                  qws, kws, vtw);
  attn_kernel<<<dim3(192, 8), 128, 0, stream>>>(qws, kws, vtw, atw);
  oproj_kernel<<<dim3(256, 12), 32, 0, stream>>>(atw, wob, wo_b, (float*)d_out);
}
